// MultiHeadAttention_103079215699
// MI455X (gfx1250) — compile-verified, matrix operand order checked by reading, not run on hardware
//
#include <hip/hip_runtime.h>
#include <hip/hip_bf16.h>

typedef __attribute__((ext_vector_type(16))) __bf16 v16bf;
typedef __attribute__((ext_vector_type(8)))  __bf16 v8bf;
typedef __attribute__((ext_vector_type(8)))  float  v8f;

#define D_MODEL   1024
#define NUM_HEADS 16
#define HEAD_DIM  64
#define SEQ       2048
#define BATCH     4

#if __has_builtin(__builtin_amdgcn_global_load_async_to_lds_b128) && \
    __has_builtin(__builtin_amdgcn_s_wait_asynccnt)
#define HAVE_ASYNC_LDS 1
typedef int b128_t __attribute__((vector_size(16)));
typedef __attribute__((address_space(1))) b128_t* global_b128_p;
typedef __attribute__((address_space(3))) b128_t* lds_b128_p;
#else
#define HAVE_ASYNC_LDS 0
#endif

// ---------------------------------------------------------------------------
// WMMA helper: D = A(16x32 bf16) * B(32x16 bf16) + C(16x16 f32)
static __device__ __forceinline__ v8f wmma_bf16(v16bf a, v16bf b, v8f c) {
    return __builtin_amdgcn_wmma_f32_16x16x32_bf16(
        /*neg_a=*/false, a, /*neg_b=*/false, b,
        /*c_mod=*/(short)0, c, /*reuse_a=*/false, /*reuse_b=*/false);
}

// Load a 16-element bf16 fragment from LDS as two 16B-aligned v8bf reads
// (row strides below are multiples of 16B, so each half is aligned).
static __device__ __forceinline__ v16bf load_frag(const __bf16* p) {
    v8bf lo = *(const v8bf*)p;
    v8bf hi = *(const v8bf*)(p + 8);
    return __builtin_shufflevector(lo, hi, 0,1,2,3,4,5,6,7,8,9,10,11,12,13,14,15);
}

// ---------------------------------------------------------------------------
// Weight pack: W fp32 [K,N] -> Wt bf16 [N,K] (transposed, L2-resident, 2MB).
__global__ void pack_weight_kernel(const float* __restrict__ W,
                                   __bf16* __restrict__ Wt, int K, int N)
{
    int idx = blockIdx.x * blockDim.x + threadIdx.x;   // over K*N
    int n = idx % N;
    int k = idx / N;
    Wt[(size_t)n * K + k] = (__bf16)W[idx];
}

// ---------------------------------------------------------------------------
// GEMM: C[M,N] = A[M,K] * Bt[N,K]^T + bias[N].  A fp32, Bt packed bf16.
// 256 threads = 8 waves; block tile 128x128, K-step 32 (one WMMA K per stage).
#define BM 128
#define BN 128
#define BK 32
#define ASTR 40   // 32 + 8 pad elems -> 80B row stride (16B aligned)
#define BSTR 40

__global__ __launch_bounds__(256)
void gemm_bt_bias_kernel(const float* __restrict__ A, const __bf16* __restrict__ Bt,
                         const float* __restrict__ bias, float* __restrict__ C,
                         int M, int N, int K)
{
    __shared__ alignas(16) __bf16 sA [BM * ASTR];   // [m][k]
    __shared__ alignas(16) __bf16 sBt[BN * BSTR];   // [n][k]

    const int tid  = threadIdx.x;
    const int lane = tid & 31;
    const int wave = tid >> 5;
    const int half = lane >> 4;
    const int l16  = lane & 15;

    const int m_wave = (wave & 3) * 32;   // wave's M offset in tile (4 rows of waves)
    const int n_wave = (wave >> 2) * 64;  // wave's N offset in tile (2 cols of waves)

    const int block_m = blockIdx.y * BM;
    const int block_n = blockIdx.x * BN;

    v8f acc[2][4] = {};

    for (int k0 = 0; k0 < K; k0 += BK) {
        __syncthreads();
        // --- B tile: 128 rows x 32 bf16 = 512 x 16B chunks, 2 per thread.
        // Packed bf16 + pre-transposed, so this is a pure contiguous copy:
        // use the gfx1250 async LDS-DMA path (ASYNCcnt) when available.
        #pragma unroll
        for (int i = 0; i < 2; ++i) {
            int ch    = tid + i * 256;            // 0..511
            int row   = ch >> 2;                  // 4 chunks per 32-elem row
            int chunk = ch & 3;
            const __bf16* gsrc = Bt + (size_t)(block_n + row) * K + k0 + chunk * 8;
            __bf16*       ldst = &sBt[row * BSTR + chunk * 8];
#if HAVE_ASYNC_LDS
            __builtin_amdgcn_global_load_async_to_lds_b128(
                (global_b128_p)gsrc, (lds_b128_p)ldst, 0, 0);
#else
            *(v8bf*)ldst = *(const v8bf*)gsrc;
#endif
        }
        // --- A tile: 128x32 floats = 1024 float4, 4 per thread (convert to bf16)
        #pragma unroll
        for (int i = 0; i < 4; ++i) {
            int e4  = tid + i * 256;            // 0..1023
            int row = e4 >> 3;                  // 8 float4 per 32-wide row
            int c4  = (e4 & 7) << 2;
            float4 f = *(const float4*)(A + (size_t)(block_m + row) * K + k0 + c4);
            __bf16* d = &sA[row * ASTR + c4];
            d[0] = (__bf16)f.x; d[1] = (__bf16)f.y;
            d[2] = (__bf16)f.z; d[3] = (__bf16)f.w;
        }
#if HAVE_ASYNC_LDS
        __builtin_amdgcn_s_wait_asynccnt(0);
#endif
        __syncthreads();

        // Fragments: lane<16 -> K 0..15, lane>=16 -> K 16..31 (half*16)
        v16bf afrag[2], bfrag[4];
        #pragma unroll
        for (int tm = 0; tm < 2; ++tm)
            afrag[tm] = load_frag(&sA[(m_wave + tm * 16 + l16) * ASTR + half * 16]);
        #pragma unroll
        for (int tn = 0; tn < 4; ++tn)
            bfrag[tn] = load_frag(&sBt[(n_wave + tn * 16 + l16) * BSTR + half * 16]);

        #pragma unroll
        for (int tm = 0; tm < 2; ++tm)
            #pragma unroll
            for (int tn = 0; tn < 4; ++tn)
                acc[tm][tn] = wmma_bf16(afrag[tm], bfrag[tn], acc[tm][tn]);
    }

    // Epilogue: C layout -> lane<16: M=r, N=l16 ; lane>=16: M=8+r
    #pragma unroll
    for (int tm = 0; tm < 2; ++tm) {
        #pragma unroll
        for (int tn = 0; tn < 4; ++tn) {
            int m_base = block_m + m_wave + tm * 16 + half * 8;
            int n      = block_n + n_wave + tn * 16 + l16;
            float bv   = bias[n];
            #pragma unroll
            for (int r = 0; r < 8; ++r)
                C[(size_t)(m_base + r) * N + n] = acc[tm][tn][r] + bv;
        }
    }
}

// ---------------------------------------------------------------------------
// RoPE, in-place; one thread per (b,s,h,j) pair handles elements d=j and d=j+32.
__global__ void rope_kernel(float* X)
{
    int idx = blockIdx.x * blockDim.x + threadIdx.x;   // over B*S*H*32
    int j   = idx & 31;
    int t   = idx >> 5;
    int h   = t & (NUM_HEADS - 1);
    t >>= 4;
    int s   = t & (SEQ - 1);
    int b   = t >> 11;

    // inv_freq = 10000^(-2j/64) = exp(-ln(10000) * 2j/64)
    float inv_freq = __expf(-9.210340372f * (float)(2 * j) * (1.0f / 64.0f));
    float ang = (float)s * inv_freq;
    float sn, cs;
    __sincosf(ang, &sn, &cs);

    size_t base = ((size_t)b * SEQ + s) * D_MODEL + h * HEAD_DIM + j;
    float x = X[base];
    float y = X[base + 32];
    X[base]      = x * cs - y * sn;
    X[base + 32] = y * cs + x * sn;
}

// ---------------------------------------------------------------------------
// Flash attention: one workgroup per (b, h, 64-query block); 4 waves x 16 rows.
#define FA_BN 64
#define KSTR  72   // 64 + 8 pad elems -> 144B row stride (16B aligned)

__global__ __launch_bounds__(128)
void flash_attn_kernel(const float* Q, const float* __restrict__ K,
                       const float* __restrict__ V, float* Octx)
{
    __shared__ alignas(16) __bf16 sK [FA_BN * KSTR];       // [n][d]
    __shared__ alignas(16) __bf16 sVt[HEAD_DIM * KSTR];    // [d][n]
    __shared__ alignas(16) __bf16 sP [4 * 16 * KSTR];      // per-wave [16][n]

    const int tid  = threadIdx.x;
    const int lane = tid & 31;
    const int wave = tid >> 5;
    const int half = lane >> 4;
    const int l16  = lane & 15;

    const int b  = blockIdx.z;
    const int h  = blockIdx.y;
    const int m0 = blockIdx.x * 64 + wave * 16;            // wave's query rows

    const size_t head_off = (size_t)b * SEQ * D_MODEL + (size_t)h * HEAD_DIM;

    // Q fragments, pre-scaled by 1/sqrt(hd)=0.125; held for the whole KV loop.
    v16bf qfrag[2];
    {
        const float* qrow = Q + head_off + (size_t)(m0 + l16) * D_MODEL;
        #pragma unroll
        for (int ks = 0; ks < 2; ++ks) {
            const float* p = qrow + ks * 32 + half * 16;
            v16bf a;
            #pragma unroll
            for (int q4 = 0; q4 < 4; ++q4) {
                float4 f = *(const float4*)(p + q4 * 4);
                a[q4 * 4 + 0] = (__bf16)(f.x * 0.125f);
                a[q4 * 4 + 1] = (__bf16)(f.y * 0.125f);
                a[q4 * 4 + 2] = (__bf16)(f.z * 0.125f);
                a[q4 * 4 + 3] = (__bf16)(f.w * 0.125f);
            }
            qfrag[ks] = a;
        }
    }

    float mrow[8], lrow[8];
    #pragma unroll
    for (int r = 0; r < 8; ++r) { mrow[r] = -3.0e38f; lrow[r] = 0.0f; }
    v8f oacc[4] = {};

    for (int kb = 0; kb < SEQ; kb += FA_BN) {
        __syncthreads();
        // K tile -> sK[n][d] and V tile -> sVt[d][n]; 64x64 = 1024 float4, 8/thread
        #pragma unroll
        for (int i = 0; i < 8; ++i) {
            int e4 = tid + i * 128;
            int n  = e4 >> 4;
            int d4 = (e4 & 15) << 2;
            float4 fk = *(const float4*)(K + head_off + (size_t)(kb + n) * D_MODEL + d4);
            __bf16* d = &sK[n * KSTR + d4];
            d[0] = (__bf16)fk.x; d[1] = (__bf16)fk.y;
            d[2] = (__bf16)fk.z; d[3] = (__bf16)fk.w;
            float4 fv = *(const float4*)(V + head_off + (size_t)(kb + n) * D_MODEL + d4);
            sVt[(d4 + 0) * KSTR + n] = (__bf16)fv.x;
            sVt[(d4 + 1) * KSTR + n] = (__bf16)fv.y;
            sVt[(d4 + 2) * KSTR + n] = (__bf16)fv.z;
            sVt[(d4 + 3) * KSTR + n] = (__bf16)fv.w;
        }
        __syncthreads();

        // S = Q * K^T : B-fragment of K^T = contiguous rows of sK
        v8f sacc[4] = {};
        #pragma unroll
        for (int ks = 0; ks < 2; ++ks)
            #pragma unroll
            for (int t = 0; t < 4; ++t) {
                v16bf bfrag = load_frag(&sK[(t * 16 + l16) * KSTR + ks * 32 + half * 16]);
                sacc[t] = wmma_bf16(qfrag[ks], bfrag, sacc[t]);
            }

        // Online softmax; each lane-half owns rows m = 8*half + r
        #pragma unroll
        for (int r = 0; r < 8; ++r) {
            float vmax = fmaxf(fmaxf(sacc[0][r], sacc[1][r]),
                               fmaxf(sacc[2][r], sacc[3][r]));
            #pragma unroll
            for (int off = 8; off >= 1; off >>= 1)
                vmax = fmaxf(vmax, __shfl_xor(vmax, off, 16));
            float mnew  = fmaxf(mrow[r], vmax);
            float alpha = __expf(mrow[r] - mnew);
            float psum  = 0.0f;
            #pragma unroll
            for (int t = 0; t < 4; ++t) {
                float p = __expf(sacc[t][r] - mnew);
                sacc[t][r] = p;
                psum += p;
            }
            #pragma unroll
            for (int off = 8; off >= 1; off >>= 1)
                psum += __shfl_xor(psum, off, 16);
            lrow[r] = lrow[r] * alpha + psum;
            mrow[r] = mnew;
            #pragma unroll
            for (int t = 0; t < 4; ++t) oacc[t][r] *= alpha;
        }

        // P (C-layout) -> LDS -> A-layout fragments
        __bf16* pbase = &sP[wave * 16 * KSTR];
        #pragma unroll
        for (int t = 0; t < 4; ++t)
            #pragma unroll
            for (int r = 0; r < 8; ++r)
                pbase[(8 * half + r) * KSTR + t * 16 + l16] = (__bf16)sacc[t][r];
        __syncthreads();

        // O += P * V  (B-fragment of V = contiguous rows of sVt)
        #pragma unroll
        for (int ks = 0; ks < 2; ++ks) {
            v16bf pfrag = load_frag(&pbase[l16 * KSTR + ks * 32 + half * 16]);
            #pragma unroll
            for (int dt = 0; dt < 4; ++dt) {
                v16bf vfrag = load_frag(&sVt[(dt * 16 + l16) * KSTR + ks * 32 + half * 16]);
                oacc[dt] = wmma_bf16(pfrag, vfrag, oacc[dt]);
            }
        }
    }

    // Normalize and store context in [B,S,D] layout (head h columns)
    #pragma unroll
    for (int r = 0; r < 8; ++r) {
        float inv_l = 1.0f / lrow[r];
        int   srow  = m0 + 8 * half + r;
        float* orow = Octx + head_off + (size_t)srow * D_MODEL;
        #pragma unroll
        for (int dt = 0; dt < 4; ++dt)
            orow[dt * 16 + l16] = oacc[dt][r] * inv_l;
    }
}

// ---------------------------------------------------------------------------
extern "C" void kernel_launch(void* const* d_in, const int* in_sizes, int n_in,
                              void* d_out, int out_size, void* d_ws, size_t ws_size,
                              hipStream_t stream)
{
    (void)in_sizes; (void)n_in; (void)out_size; (void)ws_size;

    const float* x_q = (const float*)d_in[0];
    const float* x_k = (const float*)d_in[1];
    const float* x_v = (const float*)d_in[2];
    const float* Wq  = (const float*)d_in[3];
    const float* bq  = (const float*)d_in[4];
    const float* Wk  = (const float*)d_in[5];
    const float* bk  = (const float*)d_in[6];
    const float* Wv  = (const float*)d_in[7];
    const float* bv  = (const float*)d_in[8];
    const float* Wo  = (const float*)d_in[9];
    const float* bo  = (const float*)d_in[10];
    float* out = (float*)d_out;

    const int    M  = BATCH * SEQ;             // 8192
    const size_t DD = (size_t)D_MODEL * D_MODEL;

    float*  Qb  = (float*)d_ws;                // 32 MB
    float*  Kb  = Qb + (size_t)M * D_MODEL;    // 32 MB
    float*  Vb  = Kb + (size_t)M * D_MODEL;    // 32 MB
    __bf16* Wtq = (__bf16*)(Vb + (size_t)M * D_MODEL);   // 2 MB each
    __bf16* Wtk = Wtq + DD;
    __bf16* Wtv = Wtk + DD;
    __bf16* Wto = Wtv + DD;                    // total ws: 104 MB

    // Pack weights once per launch: fp32 [K,N] -> bf16 [N,K]
    const int pack_blocks = (int)(DD / 256);
    pack_weight_kernel<<<pack_blocks, 256, 0, stream>>>(Wq, Wtq, D_MODEL, D_MODEL);
    pack_weight_kernel<<<pack_blocks, 256, 0, stream>>>(Wk, Wtk, D_MODEL, D_MODEL);
    pack_weight_kernel<<<pack_blocks, 256, 0, stream>>>(Wv, Wtv, D_MODEL, D_MODEL);
    pack_weight_kernel<<<pack_blocks, 256, 0, stream>>>(Wo, Wto, D_MODEL, D_MODEL);

    dim3 gemm_grid(D_MODEL / BN, M / BM);      // (8, 64)

    gemm_bt_bias_kernel<<<gemm_grid, 256, 0, stream>>>(x_q, Wtq, bq, Qb, M, D_MODEL, D_MODEL);
    gemm_bt_bias_kernel<<<gemm_grid, 256, 0, stream>>>(x_k, Wtk, bk, Kb, M, D_MODEL, D_MODEL);
    gemm_bt_bias_kernel<<<gemm_grid, 256, 0, stream>>>(x_v, Wtv, bv, Vb, M, D_MODEL, D_MODEL);

    const int pairs = BATCH * SEQ * NUM_HEADS * 32;   // 4M pair-threads
    rope_kernel<<<pairs / 256, 256, 0, stream>>>(Qb);
    rope_kernel<<<pairs / 256, 256, 0, stream>>>(Kb);

    // Context written in-place over Qb (each workgroup reads/writes only its own
    // (b, h, row-block) region, so the reuse is race-free).
    flash_attn_kernel<<<dim3(SEQ / 64, NUM_HEADS, BATCH), 128, 0, stream>>>(Qb, Kb, Vb, Qb);

    gemm_bt_bias_kernel<<<gemm_grid, 256, 0, stream>>>(Qb, Wto, bo, out, M, D_MODEL, D_MODEL);
}
